// EGAE_11510512353896
// MI455X (gfx1250) — compile-verified, matrix operand order checked by reading, not run on hardware
//
#include <hip/hip_runtime.h>
#include <hip/hip_bf16.h>

// ---------------------------------------------------------------------------
// EGAE forward on MI455X (gfx1250, wave32, WMMA).
// Pipeline: prep(W1^T,W2^T f16) ; zero(H1,Y3) ; gemm1(WMMA) ; spmm32(atomics)
//           ; gemm2(relu+WMMA) ; spmm16(atomics) ; rownorm -> d_out
// ---------------------------------------------------------------------------

#define N_NODES 100000
#define D_FEAT  512
#define N_EDGES 3200000
#define L1D     32
#define L2D     16

typedef __attribute__((ext_vector_type(16))) _Float16 v16h;
typedef __attribute__((ext_vector_type(8)))  _Float16 v8h;
typedef __attribute__((ext_vector_type(8)))  float    v8f;
typedef __attribute__((ext_vector_type(4)))  float    v4f;

// --- prep: transpose + f32->f16 convert the tiny weight matrices -----------
// W1 [512][32] -> W1t [32][512] f16 ; W2 [32][16] -> W2t [16][32] f16
__global__ void egae_prep(const float* __restrict__ W1,
                          const float* __restrict__ W2,
                          _Float16* __restrict__ W1t,
                          _Float16* __restrict__ W2t) {
  int t = blockIdx.x * blockDim.x + threadIdx.x;
  if (t < D_FEAT * L1D) {
    int k = t / L1D, n = t % L1D;
    W1t[(size_t)n * D_FEAT + k] = (_Float16)W1[t];
  }
  if (t < L1D * L2D) {
    int k = t / L2D, n = t % L2D;
    W2t[(size_t)n * L1D + k] = (_Float16)W2[t];
  }
}

__global__ void egae_zero(float* __restrict__ p, int n) {
  int i = blockIdx.x * blockDim.x + threadIdx.x;
  if (i < n) p[i] = 0.0f;
}

// --- gemm1: Y1 = X @ W1 ; one wave -> 16x32 tile, K=512 in steps of 32 -----
__global__ void egae_gemm1(const float* __restrict__ X,
                           const _Float16* __restrict__ W1t,
                           float* __restrict__ Y1) {
  const int lane = threadIdx.x & 31;
  const int wave = (blockIdx.x * blockDim.x + threadIdx.x) >> 5;
  const int row0 = wave << 4;
  if (row0 >= N_NODES) return;               // wave-uniform exit
  const int m  = lane & 15;                  // A row / D column within tile
  const int kh = lane >> 4;                  // K-half select (A/B layout)

  const float*    arow = X   + (size_t)(row0 + m) * D_FEAT + kh * 8;
  const _Float16* bc0  = W1t + (size_t)m         * D_FEAT + kh * 8; // col m
  const _Float16* bc1  = W1t + (size_t)(m + 16)  * D_FEAT + kh * 8; // col m+16

  v8f acc0 = {}; v8f acc1 = {};
  for (int k0 = 0; k0 < D_FEAT; k0 += 32) {
    // A fragment: two contiguous 8-float runs -> 16 f16 per lane
    v4f a0 = *(const v4f*)(arow + k0);
    v4f a1 = *(const v4f*)(arow + k0 + 4);
    v4f a2 = *(const v4f*)(arow + k0 + 16);
    v4f a3 = *(const v4f*)(arow + k0 + 20);
    v16h af;
#pragma unroll
    for (int i = 0; i < 4; ++i) {
      af[i]      = (_Float16)a0[i];
      af[4 + i]  = (_Float16)a1[i];
      af[8 + i]  = (_Float16)a2[i];
      af[12 + i] = (_Float16)a3[i];
    }
    // B fragments: pre-transposed f16 columns, contiguous 16B runs
    v8h b0lo = *(const v8h*)(bc0 + k0);
    v8h b0hi = *(const v8h*)(bc0 + k0 + 16);
    v8h b1lo = *(const v8h*)(bc1 + k0);
    v8h b1hi = *(const v8h*)(bc1 + k0 + 16);
    v16h bf0, bf1;
#pragma unroll
    for (int i = 0; i < 8; ++i) {
      bf0[i] = b0lo[i]; bf0[8 + i] = b0hi[i];
      bf1[i] = b1lo[i]; bf1[8 + i] = b1hi[i];
    }
    acc0 = __builtin_amdgcn_wmma_f32_16x16x32_f16(false, af, false, bf0,
                                                  (short)0, acc0, false, false);
    acc1 = __builtin_amdgcn_wmma_f32_16x16x32_f16(false, af, false, bf1,
                                                  (short)0, acc1, false, false);
  }
  // D layout: lane -> column (m), VGPR v -> row v + 8*kh
  float* drow = Y1 + (size_t)(row0 + kh * 8) * L1D + m;
#pragma unroll
  for (int v = 0; v < 8; ++v) {
    drow[(size_t)v * L1D]      = acc0[v];
    drow[(size_t)v * L1D + 16] = acc1[v];
  }
}

// --- spmm over 32 columns: H1[r] += val * Y1[c] (f32 atomics, L2-resident) -
__global__ void egae_spmm32(const int* __restrict__ er, const int* __restrict__ ec,
                            const float* __restrict__ ev,
                            const float* __restrict__ src, float* __restrict__ dst) {
  long long t = (long long)blockIdx.x * blockDim.x + threadIdx.x;
  int e = (int)(t >> 5);
  if (e >= N_EDGES) return;
  int col = (int)(t & 31);
  float x = ev[e] * src[(size_t)ec[e] * L1D + col];
  atomicAdd(dst + (size_t)er[e] * L1D + col, x);
}

// --- gemm2: Y2 = relu(H1) @ W2 ; one WMMA per 16 rows (K=32, N=16) ---------
__global__ void egae_gemm2(const float* __restrict__ H1,
                           const _Float16* __restrict__ W2t,
                           float* __restrict__ Y2) {
  const int lane = threadIdx.x & 31;
  const int wave = (blockIdx.x * blockDim.x + threadIdx.x) >> 5;
  const int row0 = wave << 4;
  if (row0 >= N_NODES) return;
  const int m  = lane & 15;
  const int kh = lane >> 4;

  const float* arow = H1 + (size_t)(row0 + m) * L1D + kh * 8;
  v4f a0 = *(const v4f*)(arow);
  v4f a1 = *(const v4f*)(arow + 4);
  v4f a2 = *(const v4f*)(arow + 16);
  v4f a3 = *(const v4f*)(arow + 20);
  v16h af;
#pragma unroll
  for (int i = 0; i < 4; ++i) {       // relu fused into the f16 conversion
    af[i]      = (_Float16)fmaxf(a0[i], 0.0f);
    af[4 + i]  = (_Float16)fmaxf(a1[i], 0.0f);
    af[8 + i]  = (_Float16)fmaxf(a2[i], 0.0f);
    af[12 + i] = (_Float16)fmaxf(a3[i], 0.0f);
  }
  const _Float16* bc = W2t + (size_t)m * L1D + kh * 8;
  v8h blo = *(const v8h*)(bc);
  v8h bhi = *(const v8h*)(bc + 16);
  v16h bf;
#pragma unroll
  for (int i = 0; i < 8; ++i) { bf[i] = blo[i]; bf[8 + i] = bhi[i]; }

  v8f acc = {};
  acc = __builtin_amdgcn_wmma_f32_16x16x32_f16(false, af, false, bf,
                                               (short)0, acc, false, false);
  float* drow = Y2 + (size_t)(row0 + kh * 8) * L2D + m;
#pragma unroll
  for (int v = 0; v < 8; ++v) drow[(size_t)v * L2D] = acc[v];
}

// --- spmm over 16 columns: Y3[r] += val * Y2[c] ----------------------------
__global__ void egae_spmm16(const int* __restrict__ er, const int* __restrict__ ec,
                            const float* __restrict__ ev,
                            const float* __restrict__ src, float* __restrict__ dst) {
  long long t = (long long)blockIdx.x * blockDim.x + threadIdx.x;
  int e = (int)(t >> 4);
  if (e >= N_EDGES) return;
  int col = (int)(t & 15);
  float x = ev[e] * src[(size_t)ec[e] * L2D + col];
  atomicAdd(dst + (size_t)er[e] * L2D + col, x);
}

// --- row-wise L2 normalization with eps clamp ------------------------------
__global__ void egae_norm(const float* __restrict__ Y3, float* __restrict__ out) {
  int i = blockIdx.x * blockDim.x + threadIdx.x;
  if (i >= N_NODES) return;
  const v4f* s = (const v4f*)(Y3 + (size_t)i * L2D);
  v4f x0 = s[0], x1 = s[1], x2 = s[2], x3 = s[3];
  float ss = 0.0f;
#pragma unroll
  for (int j = 0; j < 4; ++j)
    ss += x0[j] * x0[j] + x1[j] * x1[j] + x2[j] * x2[j] + x3[j] * x3[j];
  float inv = 1.0f / fmaxf(sqrtf(ss), 1e-7f);
  v4f* o = (v4f*)(out + (size_t)i * L2D);
  o[0] = x0 * inv; o[1] = x1 * inv; o[2] = x2 * inv; o[3] = x3 * inv;
}

extern "C" void kernel_launch(void* const* d_in, const int* in_sizes, int n_in,
                              void* d_out, int out_size, void* d_ws, size_t ws_size,
                              hipStream_t stream) {
  const float* X  = (const float*)d_in[0];
  const float* W1 = (const float*)d_in[1];
  const float* W2 = (const float*)d_in[2];
  const int*   lr = (const int*)d_in[3];
  const int*   lc = (const int*)d_in[4];
  const float* lv = (const float*)d_in[5];
  float* out = (float*)d_out;

  char* ws = (char*)d_ws;                       // all offsets 16B-aligned
  float*    Y1  = (float*)(ws + 0);             // 100000*32*4 = 12.8 MB
  float*    H1  = (float*)(ws + 12800000);      // 12.8 MB (accumulator)
  float*    Y2  = (float*)(ws + 25600000);      //  6.4 MB
  float*    Y3  = (float*)(ws + 32000000);      //  6.4 MB (accumulator)
  _Float16* W1t = (_Float16*)(ws + 38400000);   //  32 KB
  _Float16* W2t = (_Float16*)(ws + 38432768);   //   1 KB

  const int waves  = N_NODES / 16;              // 6250 (exact)
  const int gblk   = (waves + 7) / 8;           // 8 waves / 256-thread block

  egae_prep<<<(D_FEAT * L1D + 255) / 256, 256, 0, stream>>>(W1, W2, W1t, W2t);
  egae_zero<<<(N_NODES * L1D + 255) / 256, 256, 0, stream>>>(H1, N_NODES * L1D);
  egae_zero<<<(N_NODES * L2D + 255) / 256, 256, 0, stream>>>(Y3, N_NODES * L2D);
  egae_gemm1<<<gblk, 256, 0, stream>>>(X, W1t, Y1);
  egae_spmm32<<<(N_EDGES * 32) / 256, 256, 0, stream>>>(lr, lc, lv, Y1, H1);
  egae_gemm2<<<gblk, 256, 0, stream>>>(H1, W2t, Y2);
  egae_spmm16<<<(N_EDGES * 16) / 256, 256, 0, stream>>>(lr, lc, lv, Y2, Y3);
  egae_norm<<<(N_NODES + 255) / 256, 256, 0, stream>>>(Y3, out);
}
